// MultiTransformGinPredictorAgent_34256659153341
// MI455X (gfx1250) — matrix operand order repaired and checked by reading, not compiled, not run on hardware
//
#include <hip/hip_runtime.h>
#include <hip/hip_bf16.h>

// ---------------------------------------------------------------------------
// MultiTransformGinPredictor for MI455X (gfx1250, wave32, WMMA)
// Pipeline per transform t:
//   conv_l: zero(agg); scatter_add(h[src]->agg[dst]);    (L2-resident f32 atomics)
//           gin_mlp (WMMA f16 16x16x32): u=relu((h+agg)@waT+ba); v=u@wbT+bb;
//                    y=relu(v) stored + BN stats (LDS ds_add_f32 -> global atomics)
//           bn_finalize (scale/shift); normalize (y->h)
//   pool:   atomicAdd into emb[g, t*36+c]; count nodes per graph
// head:     256-thread blocks, one thread per graph, weights in LDS, sigmoid output
// ---------------------------------------------------------------------------

typedef _Float16 f16;
typedef __attribute__((ext_vector_type(16))) _Float16 v16h;
typedef __attribute__((ext_vector_type(8)))  float    v8f;

union Frag16 { v16h v; uint32_t u[8]; };

#define DIMC 36
#define DPAD 48   // padded N-dim (3 x 16)
#define KPAD 64   // padded K-dim (2 x 32)

// ---------------------------------------------------------------- utilities
__global__ void zero_kernel(float* __restrict__ p, int n) {
    int i = blockIdx.x * blockDim.x + threadIdx.x;
    if (i < n) p[i] = 0.f;
}

template<int D>
__global__ void scatter_kernel(const float* __restrict__ h,
                               const int* __restrict__ src,
                               const int* __restrict__ dst,
                               float* __restrict__ agg, int nEdges) {
    int idx = blockIdx.x * blockDim.x + threadIdx.x;
    int total = nEdges * D;
    if (idx >= total) return;
    int e = idx / D;
    int c = idx - e * D;
    atomicAdd(&agg[dst[e] * D + c], h[src[e] * D + c]);
}

// --------------------------------------------------- fused GIN MLP via WMMA
// y = relu( relu((h+agg) @ waT + ba) @ wbT + bb ), plus BN sum/sumsq stats.
template<int KIN>
__global__ __launch_bounds__(256)
void gin_mlp_kernel(const float* __restrict__ hin, const float* __restrict__ agg,
                    const float* __restrict__ wa,  const float* __restrict__ ba,
                    const float* __restrict__ wb,  const float* __restrict__ bb,
                    float* __restrict__ yout, float* __restrict__ stats, int nNodes) {
    __shared__ f16 A1[128 * KPAD];     // node features (K padded to 64)
    __shared__ f16 B1[DPAD * KPAD];    // waT, j-major: B1[j*64+k] = wa[j][k]
    __shared__ f16 A2[128 * KPAD];     // hidden activations
    __shared__ f16 B2[DPAD * KPAD];    // wbT
    __shared__ float sBa[DPAD], sBb[DPAD];
    __shared__ float sStat[2 * DPAD];

    const int tid  = threadIdx.x;
    const int lane = tid & 31;
    const int wave = tid >> 5;
    const int blockBase = blockIdx.x * 128;

    // zero-init LDS (padding correctness)
    uint32_t* A1u = (uint32_t*)A1; uint32_t* A2u = (uint32_t*)A2;
    uint32_t* B1u = (uint32_t*)B1; uint32_t* B2u = (uint32_t*)B2;
    for (int i = tid; i < 128 * (KPAD / 2); i += 256) { A1u[i] = 0u; A2u[i] = 0u; }
    for (int i = tid; i < DPAD * (KPAD / 2); i += 256) { B1u[i] = 0u; B2u[i] = 0u; }
    for (int i = tid; i < 2 * DPAD; i += 256) sStat[i] = 0.f;
    for (int i = tid; i < DPAD; i += 256) {
        sBa[i] = (i < DIMC) ? ba[i] : 0.f;
        sBb[i] = (i < DIMC) ? bb[i] : 0.f;
    }
    __syncthreads();

    // stage A1 = h + agg (f32 -> f16)
    for (int i = tid; i < 128 * KIN; i += 256) {
        int r = i / KIN, c = i - r * KIN;
        int node = blockBase + r;
        float v = 0.f;
        if (node < nNodes) v = hin[node * KIN + c] + agg[node * KIN + c];
        A1[r * KPAD + c] = (f16)v;
    }
    // stage B1[j][k] = wa[j][k]   (row-major [36,KIN])
    for (int i = tid; i < DIMC * KIN; i += 256) {
        int j = i / KIN, k = i - j * KIN;
        B1[j * KPAD + k] = (f16)wa[j * KIN + k];
    }
    // stage B2[j][k] = wb[j][k]   (row-major [36,36])
    for (int i = tid; i < DIMC * DIMC; i += 256) {
        int j = i / DIMC, k = i - j * DIMC;
        B2[j * KPAD + k] = (f16)wb[j * DIMC + k];
    }
    __syncthreads();

    // per-lane fragment coordinates (CDNA5 16x16x32 f16 layouts)
    const int mbase  = wave * 16;
    const int col    = lane & 15;            // B / C/D column
    const int rowA   = lane & 15;            // A row within tile
    const int hiA    = (lane >= 16) ? 8 : 0; // A K-offset half / C row half
    const int khalfB = (lane >= 16) ? 16 : 0;

    constexpr int KS1 = (KIN + 31) / 32;

    // ---- GEMM1: u = relu(A1 @ B1^T + ba), written to A2 -------------------
    for (int jt = 0; jt < 3; ++jt) {
        v8f acc = {};
        for (int ks = 0; ks < KS1; ++ks) {
            const int Koff = ks * 32;
            Frag16 a, b;
#pragma unroll
            for (int p = 0; p < 8; ++p) {
                int k0 = ((p < 4) ? 2 * p : 16 + 2 * (p - 4)) + hiA + Koff;
                a.u[p] = *(const uint32_t*)&A1[(mbase + rowA) * KPAD + k0];
                int kb = ((p < 4) ? 2 * p : 16 + 2 * (p - 4)) + hiA + Koff;
                b.u[p] = *(const uint32_t*)&B1[(jt * 16 + col) * KPAD + kb];
            }
            acc = __builtin_amdgcn_wmma_f32_16x16x32_f16(
                false, a.v, false, b.v, (short)0, acc, false, false);
        }
        const int j = jt * 16 + col;
#pragma unroll
        for (int r = 0; r < 8; ++r) {
            float u = acc[r] + sBa[j];
            u = u > 0.f ? u : 0.f;
            A2[(mbase + r + hiA) * KPAD + j] = (f16)u;   // cols 48..63 stay 0
        }
    }
    __syncthreads();

    // ---- GEMM2: y = relu(A2 @ B2^T + bb) + BN stats -----------------------
    for (int jt = 0; jt < 3; ++jt) {
        v8f acc = {};
        for (int ks = 0; ks < 2; ++ks) {
            const int Koff = ks * 32;
            Frag16 a, b;
#pragma unroll
            for (int p = 0; p < 8; ++p) {
                int k0 = ((p < 4) ? 2 * p : 16 + 2 * (p - 4)) + hiA + Koff;
                a.u[p] = *(const uint32_t*)&A2[(mbase + rowA) * KPAD + k0];
                int kb = ((p < 4) ? 2 * p : 16 + 2 * (p - 4)) + hiA + Koff;
                b.u[p] = *(const uint32_t*)&B2[(jt * 16 + col) * KPAD + kb];
            }
            acc = __builtin_amdgcn_wmma_f32_16x16x32_f16(
                false, a.v, false, b.v, (short)0, acc, false, false);
        }
        const int j = jt * 16 + col;
        if (j < DIMC) {
            float s = 0.f, q = 0.f;
#pragma unroll
            for (int r = 0; r < 8; ++r) {
                int node = blockBase + mbase + r + hiA;
                float yv = acc[r] + sBb[j];
                yv = yv > 0.f ? yv : 0.f;
                if (node < nNodes) {
                    yout[node * DIMC + j] = yv;
                    s += yv; q += yv * yv;
                }
            }
            atomicAdd(&sStat[j], s);           // ds_add_f32
            atomicAdd(&sStat[DPAD + j], q);
        }
    }
    __syncthreads();
    if (tid < DIMC) {
        atomicAdd(&stats[tid], sStat[tid]);
        atomicAdd(&stats[DPAD + tid], sStat[DPAD + tid]);
    }
}

// ---------------------------------------------------------------- batchnorm
__global__ void bn_finalize_kernel(const float* __restrict__ stats,
                                   const float* __restrict__ g,
                                   const float* __restrict__ b,
                                   float* __restrict__ scale,
                                   float* __restrict__ shift, float invN) {
    int j = threadIdx.x;
    if (j < DIMC) {
        float mean = stats[j] * invN;
        float var  = stats[DPAD + j] * invN - mean * mean;
        float sc   = g[j] * rsqrtf(var + 1e-5f);
        scale[j] = sc;
        shift[j] = b[j] - mean * sc;
    }
}

__global__ void normalize_kernel(const float* __restrict__ y,
                                 const float* __restrict__ scale,
                                 const float* __restrict__ shift,
                                 float* __restrict__ h, int total) {
    int i = blockIdx.x * blockDim.x + threadIdx.x;
    if (i >= total) return;
    int c = i % DIMC;
    h[i] = scale[c] * y[i] + shift[c];
}

// ------------------------------------------------------------------ pooling
__global__ void pool_kernel(const float* __restrict__ h, const int* __restrict__ batch,
                            float* __restrict__ emb, int colOff, int nNodes) {
    int idx = blockIdx.x * blockDim.x + threadIdx.x;
    int total = nNodes * DIMC;
    if (idx >= total) return;
    int n = idx / DIMC, c = idx - n * DIMC;
    atomicAdd(&emb[batch[n] * (2 * DIMC) + colOff + c], h[idx]);
}

__global__ void count_kernel(const int* __restrict__ batch, float* __restrict__ cnt,
                             int nNodes) {
    int i = blockIdx.x * blockDim.x + threadIdx.x;
    if (i < nNodes) atomicAdd(&cnt[batch[i]], 1.0f);
}

// --------------------------------------------------------------------- head
__global__ __launch_bounds__(256)
void head_kernel(const float* __restrict__ emb, const float* __restrict__ cnt,
                 const float* __restrict__ hw1, const float* __restrict__ hb1,
                 const float* __restrict__ hw2, const float* __restrict__ hb2,
                 const float* __restrict__ hw3, const float* __restrict__ hb3,
                 float* __restrict__ out, int nG) {
    __shared__ float W1[54 * 72], B1s[54], W2[18 * 54], B2s[18], W3[18], B3s;
    int tid = threadIdx.x;
    for (int i = tid; i < 54 * 72; i += 256) W1[i] = hw1[i];
    for (int i = tid; i < 18 * 54; i += 256) W2[i] = hw2[i];
    if (tid < 54) B1s[tid] = hb1[tid];
    if (tid < 18) { B2s[tid] = hb2[tid]; W3[tid] = hw3[tid]; }
    if (tid == 0) B3s = hb3[0];
    __syncthreads();

    int g = blockIdx.x * 256 + tid;
    if (g >= nG) return;

    float inv0 = 1.0f / fmaxf(cnt[g], 1.0f);
    float inv1 = 1.0f / fmaxf(cnt[nG + g], 1.0f);
    float e[72];
#pragma unroll
    for (int c = 0; c < 72; ++c)
        e[c] = emb[g * 72 + c] * ((c < 36) ? inv0 : inv1);

    float h1[54];
    for (int j = 0; j < 54; ++j) {
        float s = B1s[j];
        for (int c = 0; c < 72; ++c) s += W1[j * 72 + c] * e[c];
        h1[j] = s > 0.f ? s : 0.f;
    }
    float h2[18];
    for (int k = 0; k < 18; ++k) {
        float s = B2s[k];
        for (int j = 0; j < 54; ++j) s += W2[k * 54 + j] * h1[j];
        h2[k] = s > 0.f ? s : 0.f;
    }
    float o = B3s;
    for (int k = 0; k < 18; ++k) o += W3[k] * h2[k];
    out[g] = 1.0f / (1.0f + expf(-o));
}

// ------------------------------------------------------------------- driver
static inline int ceil_div(int a, int b) { return (a + b - 1) / b; }

extern "C" void kernel_launch(void* const* d_in, const int* in_sizes, int n_in,
                              void* d_out, int out_size, void* d_ws, size_t ws_size,
                              hipStream_t stream) {
    const int T = 2, IN = 6;
    const int N = in_sizes[0] / (T * IN);      // 100000
    const int E = in_sizes[1] / (T * 2);       // 1600000
    const int G = out_size;                    // 512

    const float* x     = (const float*)d_in[0];
    const int*   eidx  = (const int*)d_in[1];
    const int*   batch = (const int*)d_in[2];
    const float* w1a = (const float*)d_in[3];  const float* b1a = (const float*)d_in[4];
    const float* w1b = (const float*)d_in[5];  const float* b1b = (const float*)d_in[6];
    const float* w2a = (const float*)d_in[7];  const float* b2a = (const float*)d_in[8];
    const float* w2b = (const float*)d_in[9];  const float* b2b = (const float*)d_in[10];
    const float* w3a = (const float*)d_in[11]; const float* b3a = (const float*)d_in[12];
    const float* w3b = (const float*)d_in[13]; const float* b3b = (const float*)d_in[14];
    const float* bn_g = (const float*)d_in[15];
    const float* bn_b = (const float*)d_in[16];
    const float* hw1 = (const float*)d_in[17]; const float* hb1 = (const float*)d_in[18];
    const float* hw2 = (const float*)d_in[19]; const float* hb2 = (const float*)d_in[20];
    const float* hw3 = (const float*)d_in[21]; const float* hb3 = (const float*)d_in[22];
    float* out = (float*)d_out;

    // workspace layout (f32)
    size_t need = ((size_t)N * DIMC * 3 + 2 * DPAD + 2 * DPAD +
                   (size_t)G * 2 * DIMC + 2 * (size_t)G) * sizeof(float);
    if (ws_size < need) return;  // insufficient scratch; nothing safe to do

    float* agg   = (float*)d_ws;
    float* ybuf  = agg  + (size_t)N * DIMC;
    float* hbuf  = ybuf + (size_t)N * DIMC;
    float* stats = hbuf + (size_t)N * DIMC;   // 2*DPAD
    float* scale = stats + 2 * DPAD;          // DPAD
    float* shift = scale + DPAD;              // DPAD
    float* emb   = shift + DPAD;              // G*72
    float* cnt   = emb + (size_t)G * 2 * DIMC; // 2*G

    // zero pooled embeddings + counts (contiguous)
    {
        int n = G * 2 * DIMC + 2 * G;
        zero_kernel<<<ceil_div(n, 256), 256, 0, stream>>>(emb, n);
    }

    const int mlpBlocks = ceil_div(N, 128);

    for (int t = 0; t < T; ++t) {
        const int* src = eidx + (size_t)t * 2 * E;
        const int* dst = src + E;
        const int* bt  = batch + (size_t)t * N;

        // ---- conv1 (IN=6 -> 36 -> 36) ----
        zero_kernel<<<ceil_div(N * IN, 256), 256, 0, stream>>>(agg, N * IN);
        zero_kernel<<<1, 2 * DPAD, 0, stream>>>(stats, 2 * DPAD);
        scatter_kernel<6><<<ceil_div(E * 6, 256), 256, 0, stream>>>(
            x + (size_t)t * N * IN, src, dst, agg, E);
        gin_mlp_kernel<6><<<mlpBlocks, 256, 0, stream>>>(
            x + (size_t)t * N * IN, agg,
            w1a + (size_t)t * DIMC * IN,   b1a + (size_t)t * DIMC,
            w1b + (size_t)t * DIMC * DIMC, b1b + (size_t)t * DIMC,
            ybuf, stats, N);
        bn_finalize_kernel<<<1, 64, 0, stream>>>(
            stats, bn_g + (size_t)(t * 3 + 0) * DIMC, bn_b + (size_t)(t * 3 + 0) * DIMC,
            scale, shift, 1.0f / (float)N);
        normalize_kernel<<<ceil_div(N * DIMC, 256), 256, 0, stream>>>(
            ybuf, scale, shift, hbuf, N * DIMC);

        // ---- conv2 / conv3 (36 -> 36 -> 36) ----
        const float* WA[2] = { w2a + (size_t)t * DIMC * DIMC, w3a + (size_t)t * DIMC * DIMC };
        const float* BA[2] = { b2a + (size_t)t * DIMC,        b3a + (size_t)t * DIMC };
        const float* WB[2] = { w2b + (size_t)t * DIMC * DIMC, w3b + (size_t)t * DIMC * DIMC };
        const float* BB[2] = { b2b + (size_t)t * DIMC,        b3b + (size_t)t * DIMC };
        for (int l = 0; l < 2; ++l) {
            zero_kernel<<<ceil_div(N * DIMC, 256), 256, 0, stream>>>(agg, N * DIMC);
            zero_kernel<<<1, 2 * DPAD, 0, stream>>>(stats, 2 * DPAD);
            scatter_kernel<DIMC><<<ceil_div(E * DIMC, 256), 256, 0, stream>>>(
                hbuf, src, dst, agg, E);
            gin_mlp_kernel<DIMC><<<mlpBlocks, 256, 0, stream>>>(
                hbuf, agg, WA[l], BA[l], WB[l], BB[l], ybuf, stats, N);
            bn_finalize_kernel<<<1, 64, 0, stream>>>(
                stats, bn_g + (size_t)(t * 3 + l + 1) * DIMC,
                bn_b + (size_t)(t * 3 + l + 1) * DIMC,
                scale, shift, 1.0f / (float)N);
            normalize_kernel<<<ceil_div(N * DIMC, 256), 256, 0, stream>>>(
                ybuf, scale, shift, hbuf, N * DIMC);
        }

        // ---- mean pool ----
        pool_kernel<<<ceil_div(N * DIMC, 256), 256, 0, stream>>>(hbuf, bt, emb, t * DIMC, N);
        count_kernel<<<ceil_div(N, 256), 256, 0, stream>>>(bt, cnt + (size_t)t * G, N);
    }

    head_kernel<<<ceil_div(G, 256), 256, 0, stream>>>(
        emb, cnt, hw1, hb1, hw2, hb2, hw3, hb3, out, G);
}
